// ShrdMHAttention_42099269436157
// MI455X (gfx1250) — matrix operand order repaired and checked by reading, compiled, not run on hardware
//
#include <hip/hip_runtime.h>
#include <stdint.h>

#define SEQ    4096
#define DMODEL 2048
#define NHEAD  16
#define DQK    128
#define DV     128
#define DROPE  64

#define AS1 __attribute__((address_space(1)))
#define AS3 __attribute__((address_space(3)))

typedef __attribute__((ext_vector_type(16))) __bf16 bf16x16;
typedef __attribute__((ext_vector_type(8)))  __bf16 bf16x8;
typedef __attribute__((ext_vector_type(8)))  float  f32x8;
typedef __attribute__((ext_vector_type(4)))  int    i32x4;

static __device__ __forceinline__ unsigned short f32_to_bf16(float f) {
    union { float f; unsigned u; } c; c.f = f;
    unsigned u = c.u;
    u += 0x7fffu + ((u >> 16) & 1u);
    return (unsigned short)(u >> 16);
}

static __device__ __forceinline__ f32x8 zero8() {
    f32x8 v;
#pragma unroll
    for (int i = 0; i < 8; ++i) v[i] = 0.0f;
    return v;
}

// Build a 16xbf16 WMMA operand from two contiguous 8-element (16B) chunks.
static __device__ __forceinline__ bf16x16 ldfrag(const unsigned short* p0, const unsigned short* p1) {
    bf16x8 lo = *(const bf16x8*)p0;
    bf16x8 hi = *(const bf16x8*)p1;
    bf16x16 r;
#pragma unroll
    for (int i = 0; i < 8; ++i) { r[i] = lo[i]; r[i + 8] = hi[i]; }
    return r;
}

// ---- CDNA5 async global->LDS copy (ASYNCcnt path), 16 bytes per lane ----
static __device__ __forceinline__ void async_b128(const unsigned short* gsrc, unsigned short* lds_dst) {
#if __has_builtin(__builtin_amdgcn_global_load_async_to_lds_b128)
    __builtin_amdgcn_global_load_async_to_lds_b128((AS1 i32x4*)gsrc, (AS3 i32x4*)lds_dst, 0, 0);
#else
    unsigned loff = (unsigned)(size_t)(AS3 void*)lds_dst;
    asm volatile("global_load_async_to_lds_b128 %0, %1, off"
                 :: "v"(loff), "v"(gsrc) : "memory");
#endif
}

static __device__ __forceinline__ void wait_async0() {
#if __has_builtin(__builtin_amdgcn_s_wait_asynccnt)
    __builtin_amdgcn_s_wait_asynccnt(0);
#else
    asm volatile("s_wait_asynccnt 0x0" ::: "memory");
#endif
}

// Stage a 128-row x 32-element bf16 slab (8KB) into LDS: 512 x B128, 2 per thread.
static __device__ __forceinline__ void stage_slab_128x32(const unsigned short* __restrict__ gbase,
                                                         long row_stride_elems,
                                                         unsigned short* lds_base, int tid) {
#pragma unroll
    for (int i = tid; i < 512; i += 256) {
        int row = i >> 2, part = (i & 3) * 8;
        async_b128(gbase + (long)row * row_stride_elems + part, lds_base + row * 32 + part);
    }
}

// ---------------- conversion kernels ----------------

__global__ void cvt_bf16_kernel(const float* __restrict__ src, unsigned short* __restrict__ dst, int n) {
    int i = blockIdx.x * blockDim.x + threadIdx.x;
    if (i < n) dst[i] = f32_to_bf16(src[i]);
}

// src: (H, D, E) f32 -> dst: (H, E, D) bf16
__global__ void cvt_transpose_kernel(const float* __restrict__ src, unsigned short* __restrict__ dst,
                                     int H, int D, int E) {
    long i = (long)blockIdx.x * blockDim.x + threadIdx.x;
    long n = (long)H * D * E;
    if (i >= n) return;
    int e = (int)(i % E);
    long t = i / E;
    int d = (int)(t % D);
    int h = (int)(t / D);
    dst[((long)h * E + e) * D + d] = f32_to_bf16(src[i]);
}

// ---------------- projection (+RoPE) ----------------
// xb: (SEQ, DMODEL) bf16
// wt: (NHEAD, 128, DMODEL) bf16   (transposed: out-col major, K contiguous)
// out (do_rope=1): (NHEAD, SEQ, 128) bf16 row-major [rq / rk]
// out (do_rope=0): (NHEAD, 128, SEQ) bf16 transposed [vs^T]
__global__ __launch_bounds__(256) void proj_kernel(const unsigned short* __restrict__ xb,
                                                   const unsigned short* __restrict__ wt,
                                                   unsigned short* __restrict__ out,
                                                   int do_rope,
                                                   const float* __restrict__ theta_p) {
    __shared__ __align__(16) unsigned short wls[2][128 * 32];  // [col][k], double-buffered
    const int tid  = threadIdx.x;
    const int lane = tid & 31;
    const int wid  = tid >> 5;
    const int head = blockIdx.y;
    const int row0 = (blockIdx.x * 8 + wid) * 16;
    const int lh   = lane >> 4;
    const int l16  = lane & 15;

    const unsigned short* xrow  = xb + (long)(row0 + l16) * DMODEL;
    const unsigned short* wbase = wt + (long)head * 128 * DMODEL;

    f32x8 acc[8];
#pragma unroll
    for (int j = 0; j < 8; ++j) acc[j] = zero8();

    stage_slab_128x32(wbase, DMODEL, &wls[0][0], tid);
    wait_async0();
    __syncthreads();

    for (int kc = 0; kc < DMODEL / 32; ++kc) {
        const int buf = kc & 1;
        if (kc + 1 < DMODEL / 32)
            stage_slab_128x32(wbase + (kc + 1) * 32, DMODEL, &wls[buf ^ 1][0], tid);

        const int k0 = kc * 32;
        bf16x16 a = ldfrag(xrow + k0 + lh * 16, xrow + k0 + lh * 16 + 8);
#pragma unroll
        for (int j = 0; j < 8; ++j) {
            const unsigned short* wr = &wls[buf][(j * 16 + l16) * 32 + lh * 16];
            bf16x16 b = ldfrag(wr, wr + 8);
            acc[j] = __builtin_amdgcn_wmma_f32_16x16x32_bf16(false, a, false, b, (short)0, acc[j], false, false);
        }
        wait_async0();
        __syncthreads();
    }

    if (do_rope) {
        const float theta = theta_p[0];
        const float inv = 0.29730177875068026f;  // 128^-0.25
#pragma unroll
        for (int j = 0; j < 4; ++j) {
            int col = j * 16 + l16;
            float rate = -theta * (float)col / (float)DROPE;
#pragma unroll
            for (int r = 0; r < 8; ++r) {
                int row = row0 + r + lh * 8;
                float rot = (float)row * rate;
                float s = __sinf(rot), c = __cosf(rot);
                float x1 = acc[j][r], x2 = acc[j + 4][r];
                acc[j][r]     = (c * x1 - s * x2) * inv;
                acc[j + 4][r] = (s * x1 + c * x2) * inv;
            }
        }
        unsigned short* ob = out + ((long)head * SEQ + row0) * 128;
#pragma unroll
        for (int j = 0; j < 8; ++j)
#pragma unroll
            for (int r = 0; r < 8; ++r)
                ob[(long)(r + lh * 8) * 128 + j * 16 + l16] = f32_to_bf16(acc[j][r]);
    } else {
        unsigned short* ob = out + (long)head * 128 * SEQ;
#pragma unroll
        for (int j = 0; j < 8; ++j)
#pragma unroll
            for (int r = 0; r < 8; ++r)
                ob[(long)(j * 16 + l16) * SEQ + row0 + r + lh * 8] = f32_to_bf16(acc[j][r]);
    }
}

// ---------------- flash attention (K/V staged in LDS via async copies) ----------------
// rq, rk: (NHEAD, SEQ, 128) bf16 ; vst: (NHEAD, 128, SEQ) bf16 ; z: (NHEAD, SEQ, 128) bf16
__global__ __launch_bounds__(256) void attn_kernel(const unsigned short* __restrict__ rq,
                                                   const unsigned short* __restrict__ rk,
                                                   const unsigned short* __restrict__ vst,
                                                   unsigned short* __restrict__ z) {
    __shared__ __align__(16) unsigned short kls[2][32 * 128];  // [key][dim]
    __shared__ __align__(16) unsigned short vls[2][128 * 32];  // [vcol][key]
    __shared__ __align__(16) unsigned short pbuf[8][16 * 32];
    const int tid  = threadIdx.x;
    const int lane = tid & 31;
    const int wid  = tid >> 5;
    const int head = blockIdx.y;
    const int row0 = (blockIdx.x * 8 + wid) * 16;
    const int lh   = lane >> 4;
    const int l16  = lane & 15;

    const unsigned short* qrow  = rq  + ((long)head * SEQ + row0 + l16) * 128;
    const unsigned short* kbase = rk  + (long)head * SEQ * 128;
    const unsigned short* vbase = vst + (long)head * 128 * SEQ;

    f32x8 zacc[8];
#pragma unroll
    for (int j = 0; j < 8; ++j) zacc[j] = zero8();
    float m[8], l[8];
#pragma unroll
    for (int r = 0; r < 8; ++r) { m[r] = -3.0e38f; l[r] = 0.0f; }

    // Preload query A-fragments (16x128 = 4 chunks of K=32), reused every key block.
    bf16x16 afq[4];
#pragma unroll
    for (int kc = 0; kc < 4; ++kc)
        afq[kc] = ldfrag(qrow + kc * 32 + lh * 16, qrow + kc * 32 + lh * 16 + 8);

    // Block-uniform key-block count (last wave's causal range); extra fully-masked
    // blocks are inert in the online softmax (alpha=1, p=0).
    const int row0_last = (blockIdx.x * 8 + 7) * 16;
    const int nkb = (row0_last + 16 + 31) >> 5;

    // stage key block 0
#pragma unroll
    for (int i = tid; i < 512; i += 256) {
        int key = i >> 4, part = (i & 15) * 8;
        async_b128(kbase + (long)key * 128 + part, &kls[0][key * 128 + part]);
    }
    stage_slab_128x32(vbase, SEQ, &vls[0][0], tid);
    wait_async0();
    __syncthreads();

    for (int kb = 0; kb < nkb; ++kb) {
        const int buf = kb & 1;
        if (kb + 1 < nkb) {
            const int kn = (kb + 1) * 32;
#pragma unroll
            for (int i = tid; i < 512; i += 256) {
                int key = i >> 4, part = (i & 15) * 8;
                async_b128(kbase + (long)(kn + key) * 128 + part, &kls[buf ^ 1][key * 128 + part]);
            }
            stage_slab_128x32(vbase + kn, SEQ, &vls[buf ^ 1][0], tid);
        }

        const int kcol0 = kb * 32;
        f32x8 s0 = zero8(), s1 = zero8();
#pragma unroll
        for (int kc = 0; kc < 4; ++kc) {
            const unsigned short* kr0 = &kls[buf][l16 * 128 + kc * 32 + lh * 16];
            const unsigned short* kr1 = &kls[buf][(16 + l16) * 128 + kc * 32 + lh * 16];
            bf16x16 b0 = ldfrag(kr0, kr0 + 8);
            bf16x16 b1 = ldfrag(kr1, kr1 + 8);
            s0 = __builtin_amdgcn_wmma_f32_16x16x32_bf16(false, afq[kc], false, b0, (short)0, s0, false, false);
            s1 = __builtin_amdgcn_wmma_f32_16x16x32_bf16(false, afq[kc], false, b1, (short)0, s1, false, false);
        }

        // causal mask + per-row max (reduce across 16 lanes of each half)
        float rm[8];
#pragma unroll
        for (int r = 0; r < 8; ++r) {
            int row = row0 + r + lh * 8;
            int c0 = kcol0 + l16, c1 = kcol0 + 16 + l16;
            float v0 = (c0 <= row) ? s0[r] : -1.0e9f;
            float v1 = (c1 <= row) ? s1[r] : -1.0e9f;
            s0[r] = v0; s1[r] = v1;
            float t = fmaxf(v0, v1);
            t = fmaxf(t, __shfl_xor(t, 1));
            t = fmaxf(t, __shfl_xor(t, 2));
            t = fmaxf(t, __shfl_xor(t, 4));
            t = fmaxf(t, __shfl_xor(t, 8));
            rm[r] = t;
        }

        // online softmax update; write P tile (16x32 row-major) to this wave's LDS slab
#pragma unroll
        for (int r = 0; r < 8; ++r) {
            float nm = fmaxf(m[r], rm[r]);
            float alpha = __expf(m[r] - nm);
            m[r] = nm;
            float p0 = __expf(s0[r] - nm);
            float p1 = __expf(s1[r] - nm);
            float ps = p0 + p1;
            ps += __shfl_xor(ps, 1);
            ps += __shfl_xor(ps, 2);
            ps += __shfl_xor(ps, 4);
            ps += __shfl_xor(ps, 8);
            l[r] = l[r] * alpha + ps;
#pragma unroll
            for (int j = 0; j < 8; ++j) zacc[j][r] *= alpha;
            int row = r + lh * 8;
            pbuf[wid][row * 32 + l16]      = f32_to_bf16(p0);
            pbuf[wid][row * 32 + 16 + l16] = f32_to_bf16(p1);
        }
        asm volatile("" ::: "memory");  // keep LDS store->load order (DS is in-order per wave)

        // P (A frag from LDS) x V (B frags from LDS)
        const unsigned short* pr = &pbuf[wid][l16 * 32];
        bf16x16 ap = ldfrag(pr + lh * 16, pr + lh * 16 + 8);
#pragma unroll
        for (int j = 0; j < 8; ++j) {
            const unsigned short* vr = &vls[buf][(j * 16 + l16) * 32 + lh * 16];
            bf16x16 b = ldfrag(vr, vr + 8);
            zacc[j] = __builtin_amdgcn_wmma_f32_16x16x32_bf16(false, ap, false, b, (short)0, zacc[j], false, false);
        }
        wait_async0();
        __syncthreads();
    }

    unsigned short* zb = z + ((long)head * SEQ + row0) * 128;
#pragma unroll
    for (int j = 0; j < 8; ++j)
#pragma unroll
        for (int r = 0; r < 8; ++r)
            zb[(long)(r + lh * 8) * 128 + j * 16 + l16] = f32_to_bf16(zacc[j][r] / l[r]);
}

// ---------------- output projection (O^T staged in LDS) ----------------
// z: (NHEAD, SEQ, 128) bf16 ; ot: (NHEAD, DMODEL, 128) bf16 ; out: (SEQ, DMODEL) f32
__global__ __launch_bounds__(256) void outproj_kernel(const unsigned short* __restrict__ z,
                                                      const unsigned short* __restrict__ ot,
                                                      float* __restrict__ out) {
    __shared__ __align__(16) unsigned short ols[2][128 * 32];  // [col][k]
    const int tid  = threadIdx.x;
    const int lane = tid & 31;
    const int wid  = tid >> 5;
    const int row0 = (blockIdx.x * 8 + wid) * 16;
    const int col0 = blockIdx.y * 128;
    const int lh   = lane >> 4;
    const int l16  = lane & 15;

    f32x8 acc[8];
#pragma unroll
    for (int j = 0; j < 8; ++j) acc[j] = zero8();

    // chunk c = h*4 + kc over (head, K/32)
    const unsigned short* g0 = ot + (long)(col0) * 128;  // + h*DMODEL*128 + kc*32
    stage_slab_128x32(g0, 128, &ols[0][0], tid);
    wait_async0();
    __syncthreads();

    for (int c = 0; c < NHEAD * 4; ++c) {
        const int buf = c & 1;
        if (c + 1 < NHEAD * 4) {
            int hn = (c + 1) >> 2, kcn = (c + 1) & 3;
            const unsigned short* gn = ot + (long)hn * DMODEL * 128 + (long)col0 * 128 + kcn * 32;
            stage_slab_128x32(gn, 128, &ols[buf ^ 1][0], tid);
        }

        const int h = c >> 2, kc = c & 3;
        const unsigned short* zrow = z + ((long)h * SEQ + row0 + l16) * 128 + kc * 32;
        bf16x16 a = ldfrag(zrow + lh * 16, zrow + lh * 16 + 8);
#pragma unroll
        for (int j = 0; j < 8; ++j) {
            const unsigned short* orow = &ols[buf][(j * 16 + l16) * 32 + lh * 16];
            bf16x16 b = ldfrag(orow, orow + 8);
            acc[j] = __builtin_amdgcn_wmma_f32_16x16x32_bf16(false, a, false, b, (short)0, acc[j], false, false);
        }
        wait_async0();
        __syncthreads();
    }

    const float scale = 1.0f / 2048.0f;  // 1/sqrt(16*128*2048)
    float* op = out + (long)row0 * DMODEL + col0;
#pragma unroll
    for (int j = 0; j < 8; ++j)
#pragma unroll
        for (int r = 0; r < 8; ++r)
            op[(long)(r + lh * 8) * DMODEL + j * 16 + l16] = acc[j][r] * scale;
}

// ---------------- launch ----------------

extern "C" void kernel_launch(void* const* d_in, const int* in_sizes, int n_in,
                              void* d_out, int out_size, void* d_ws, size_t ws_size,
                              hipStream_t stream) {
    const float* x  = (const float*)d_in[0];
    const float* q  = (const float*)d_in[1];
    const float* k  = (const float*)d_in[2];
    const float* v  = (const float*)d_in[3];
    const float* o  = (const float*)d_in[4];
    const float* th = (const float*)d_in[5];

    unsigned short* xb  = (unsigned short*)d_ws;                         // SEQ*DMODEL
    unsigned short* wqt = xb  + (size_t)SEQ * DMODEL;
    unsigned short* wkt = wqt + (size_t)NHEAD * 128 * DMODEL;
    unsigned short* wvt = wkt + (size_t)NHEAD * 128 * DMODEL;
    unsigned short* ott = wvt + (size_t)NHEAD * 128 * DMODEL;
    unsigned short* rqb = ott + (size_t)NHEAD * DMODEL * 128;
    unsigned short* rkb = rqb + (size_t)NHEAD * SEQ * 128;
    unsigned short* vst = rkb + (size_t)NHEAD * SEQ * 128;
    unsigned short* zb  = vst + (size_t)NHEAD * 128 * SEQ;

    {
        int n = SEQ * DMODEL;
        cvt_bf16_kernel<<<(n + 255) / 256, 256, 0, stream>>>(x, xb, n);
    }
    {
        int n = NHEAD * DMODEL * 128;
        cvt_transpose_kernel<<<(n + 255) / 256, 256, 0, stream>>>(q, wqt, NHEAD, DMODEL, 128);
        cvt_transpose_kernel<<<(n + 255) / 256, 256, 0, stream>>>(k, wkt, NHEAD, DMODEL, 128);
        cvt_transpose_kernel<<<(n + 255) / 256, 256, 0, stream>>>(v, wvt, NHEAD, DMODEL, 128);
        cvt_transpose_kernel<<<(n + 255) / 256, 256, 0, stream>>>(o, ott, NHEAD, 128, DMODEL);
    }

    dim3 gp(SEQ / 128, NHEAD);
    proj_kernel<<<gp, 256, 0, stream>>>(xb, wqt, rqb, 1, th);
    proj_kernel<<<gp, 256, 0, stream>>>(xb, wkt, rkb, 1, th);
    proj_kernel<<<gp, 256, 0, stream>>>(xb, wvt, vst, 0, th);

    attn_kernel<<<dim3(SEQ / 128, NHEAD), 256, 0, stream>>>(rqb, rkb, vst, zb);

    outproj_kernel<<<dim3(SEQ / 128, DMODEL / 128), 256, 0, stream>>>(zb, ott, (float*)d_out);
}
